// MultiHeadAttention_16801912062309
// MI455X (gfx1250) — matrix operand order repaired and checked by reading, not compiled, not run on hardware
//
#include <hip/hip_runtime.h>
#include <hip/hip_bf16.h>

// Problem constants (from reference: B, N, H, E = 16, 512, 8, 128)
#define BB   16
#define NNq  512
#define HHn  8
#define EEn  128
#define HE   (HHn * EEn)      // 1024
#define NEGV (-1000000000.0f)

typedef __attribute__((ext_vector_type(16))) _Float16 v16h;
typedef __attribute__((ext_vector_type(8)))  _Float16 v8h;
typedef __attribute__((ext_vector_type(8)))  float    v8f;
typedef __attribute__((ext_vector_type(4)))  unsigned int u32x4;
typedef __attribute__((ext_vector_type(8)))  int      i32x8;
typedef __attribute__((ext_vector_type(4)))  int      i32x4;

// ---------------------------------------------------------------------------
// WMMA fragment helpers (wave32; this code's A/B k order is contiguous, k = 16*(L>=16) + s, same in both)
//
// A (16x32 f16, M x K): lane L holds row m = L&15.
//   VGPR0..3 = K (L>=16 ? 16 : 0) + 0..7    -> contiguous 16B
//   VGPR4..7 = K (L>=16 ? 16 : 0) + 8..15   -> contiguous 16B
// B (32x16 f16, K x N): lane L holds col n = L&15,
//   VGPR0..7 = K (L>=16 ? 16 : 0) + 0..15   -> contiguous 32B
//   (load from a [n][k] row-major "B-transposed" buffer)
// C/D (16x16 f32): lane L holds col n = L&15; VGPR r holds row m = r + 8*(L>=16)
// ---------------------------------------------------------------------------
static __device__ __forceinline__ v16h frag_a_load(const _Float16* tile, int ld) {
  const int lane = threadIdx.x & 31;
  const int m    = lane & 15;
  const int kb   = (lane >> 4) << 4;            // 0 or 16
  const _Float16* p = tile + (size_t)m * ld + kb;
  union { v16h v; v8h h[2]; } u;
  u.h[0] = *(const v8h*)(p);
  u.h[1] = *(const v8h*)(p + 8);
  return u.v;
}

static __device__ __forceinline__ v16h frag_b_load(const _Float16* bt, int ld) {
  const int lane = threadIdx.x & 31;
  const int n    = lane & 15;
  const int kb   = (lane >> 4) << 4;            // 0 or 16
  return *(const v16h*)(bt + (size_t)n * ld + kb);
}

static __device__ __forceinline__ v8f wmma_f16(v16h a, v16h b, v8f c) {
  // D = A*B + C, f32 accumulate; emits v_wmma_f32_16x16x32_f16
  return __builtin_amdgcn_wmma_f32_16x16x32_f16(false, a, false, b,
                                                (short)0, c, false, false);
}

// per-row reductions over the 16 lanes of each half-wave (cols of C-frag)
static __device__ __forceinline__ float rowmax16(float v) {
  v = fmaxf(v, __shfl_xor(v, 1));
  v = fmaxf(v, __shfl_xor(v, 2));
  v = fmaxf(v, __shfl_xor(v, 4));
  v = fmaxf(v, __shfl_xor(v, 8));
  return v;
}
static __device__ __forceinline__ float rowsum16(float v) {
  v += __shfl_xor(v, 1);
  v += __shfl_xor(v, 2);
  v += __shfl_xor(v, 4);
  v += __shfl_xor(v, 8);
  return v;
}

// ---------------------------------------------------------------------------
// Tensor Data Mover: 2-D tile load, global f16 -> LDS.
// Builds D# group0/group1 per CDNA5 ISA:
//   group0: [1:0]=count=1, [63:32]=lds byte addr, [120:64]=global addr,
//           [127:126]=type=2
//   group1: [17:16]=data_size=1 (2B), [79:48]=tensor_dim0, [111:80]=tensor_dim1,
//           [127:112]=tile_dim0, [143:128]=tile_dim1, [207:160]=dim0 stride
// Tracked with TENSORcnt; caller must s_wait_tensorcnt + barrier.
// ---------------------------------------------------------------------------
static __device__ __forceinline__ void tdm_load_2d_f16(
    const _Float16* gaddr, unsigned lds_byte_off,
    int tile_d0, int tile_d1, int tensor_d0, int tensor_d1,
    unsigned long long stride0_elems) {
#if __has_builtin(__builtin_amdgcn_tensor_load_to_lds)
  const unsigned long long ga = (unsigned long long)(uintptr_t)gaddr;
  u32x4 g0;
  g0.x = 1u;                                         // count=1, user descriptor
  g0.y = lds_byte_off;                               // lds_addr (bytes)
  g0.z = (unsigned)(ga & 0xffffffffull);             // global_addr[31:0]
  g0.w = (unsigned)((ga >> 32) & 0x1ffffffull) | (2u << 30); // [56:32] | type=2
  i32x8 g1;
  g1[0] = (1 << 16);                                 // data_size=1 (2 bytes)
  g1[1] = (tensor_d0 & 0xffff) << 16;                // tensor_dim0[15:0]
  g1[2] = ((tensor_d0 >> 16) & 0xffff) | ((tensor_d1 & 0xffff) << 16);
  g1[3] = ((tensor_d1 >> 16) & 0xffff) | ((tile_d0 & 0xffff) << 16);
  g1[4] = (tile_d1 & 0xffff);                        // tile_dim1 (tile_dim2=0)
  g1[5] = (int)(stride0_elems & 0xffffffffull);      // tensor_dim0_stride[31:0]
  g1[6] = (int)((stride0_elems >> 32) & 0xffffull);  // stride[47:32]; stride1=0
  g1[7] = 0;
  i32x4 z4 = {0, 0, 0, 0};
#if __clang_major__ >= 23
  i32x8 z8 = {0, 0, 0, 0, 0, 0, 0, 0};
  __builtin_amdgcn_tensor_load_to_lds(g0, g1, z4, z4, z8, 0);
#else
  __builtin_amdgcn_tensor_load_to_lds(g0, g1, z4, z4, 0);
#endif
#endif
}

// ---------------------------------------------------------------------------
// Kernel 0: f32 -> f16 conversion
// ---------------------------------------------------------------------------
__global__ void __launch_bounds__(256) cvt_f32_f16(const float* __restrict__ s,
                                                   _Float16* __restrict__ d, int n) {
  int i = blockIdx.x * blockDim.x + threadIdx.x;
  if (i < n) d[i] = (_Float16)s[i];
}

// ---------------------------------------------------------------------------
// Kernel 1: fused Q/K/V projection  (x @ W^T + b) * mask [* scale]
//   xh : (B*N, E) f16 row-major
//   wh : (HE, E)  f16 row-major  (== B-transposed layout for x @ W^T)
//   dst: TRANSP==0 -> (B,H,N,E) f16;  TRANSP==1 -> (B,H,E,N) f16
// Templated on TRANSP so the store loop is branch-free.
// ---------------------------------------------------------------------------
template <int TRANSP>
__global__ void __launch_bounds__(256, 1)
qkv_proj_kernel(const _Float16* __restrict__ xh,
                const _Float16* __restrict__ wh,
                const float* __restrict__ bias,
                const float* __restrict__ mask,
                _Float16* __restrict__ dst,
                float scale) {
  const int w    = threadIdx.x >> 5;
  const int lane = threadIdx.x & 31;
  const int nn   = lane & 15;
  const int hi   = lane >> 4;
  const int rowBase = (blockIdx.x * 8 + w) * 16;
  const int b  = rowBase / NNq;
  const int n0 = rowBase % NNq;

  v16h a[4];
#pragma unroll
  for (int t = 0; t < 4; ++t)
    a[t] = frag_a_load(xh + (size_t)rowBase * EEn + t * 32, EEn);

  float mrow[8];
#pragma unroll
  for (int r = 0; r < 8; ++r)
    mrow[r] = mask[b * NNq + n0 + r + 8 * hi] * scale;

  for (int ct = 0; ct < HE / 16; ++ct) {
    v8f acc = {};
#pragma unroll
    for (int t = 0; t < 4; ++t)
      acc = wmma_f16(a[t], frag_b_load(wh + (size_t)(ct * 16) * EEn + t * 32, EEn), acc);

    const int   col  = ct * 16 + nn;
    const int   hsel = col >> 7;       // head
    const int   e    = col & (EEn - 1);
    const float bs   = bias[col];
#pragma unroll
    for (int r = 0; r < 8; ++r) {
      const int   m   = r + 8 * hi;
      const float val = (acc[r] + bs) * mrow[r];
      if (TRANSP == 0)
        dst[(((size_t)(b * HHn + hsel)) * NNq + (n0 + m)) * EEn + e] = (_Float16)val;
      else
        dst[(((size_t)(b * HHn + hsel)) * EEn + e) * NNq + (n0 + m)] = (_Float16)val;
    }
  }
}

// ---------------------------------------------------------------------------
// Kernel 2: fused attention, flash-style online softmax.
//   grid = B*H*(N/128) blocks, 256 threads (8 waves), wave = 16 query rows.
//   All 8 waves share (b,h): K/V tiles staged once per block into LDS by the
//   Tensor Data Mover (TENSORcnt), then consumed by per-wave WMMAs.
//   __launch_bounds__(256, 1): one workgroup per WGP -> full VGPR budget,
//   no scratch spills (kernel is WMMA-bound, 512 blocks keep the chip busy).
// ---------------------------------------------------------------------------
struct AttnSmem {
  _Float16 k[32 * EEn];       // [key 0..31][e 0..127]
  _Float16 v[EEn * 32];       // [e 0..127][key 0..31]
  _Float16 p[8][16 * 32];     // per-wave P tile (C->A relayout)
};

__global__ void __launch_bounds__(256, 1)
attn_kernel(const _Float16* __restrict__ q,
            const _Float16* __restrict__ k,
            const _Float16* __restrict__ vT,
            const float* __restrict__ dist,
            const float* __restrict__ mask,
            _Float16* __restrict__ y) {
  __shared__ alignas(128) AttnSmem smem;

  const int w    = threadIdx.x >> 5;
  const int lane = threadIdx.x & 31;
  const int nn   = lane & 15;
  const int hi   = lane >> 4;
  const int qb   = blockIdx.x & 3;                  // N/128 = 4 query blocks
  const int h    = (blockIdx.x >> 2) & (HHn - 1);
  const int b    = blockIdx.x >> 5;                 // / (4*H)
  const int qrow0 = qb * 128 + w * 16;

  const _Float16* qp = q  + (((size_t)(b * HHn + h)) * NNq + qrow0) * EEn;
  const _Float16* kp = k  + ((size_t)(b * HHn + h)) * NNq * EEn;
  const _Float16* vp = vT + ((size_t)(b * HHn + h)) * EEn * NNq;

  v16h aq[4];
#pragma unroll
  for (int t = 0; t < 4; ++t) aq[t] = frag_a_load(qp + t * 32, EEn);

  float mq[8];
#pragma unroll
  for (int r = 0; r < 8; ++r) mq[r] = mask[b * NNq + qrow0 + r + 8 * hi];

  float mrow[8], lrow[8];
#pragma unroll
  for (int r = 0; r < 8; ++r) { mrow[r] = -3.0e38f; lrow[r] = 0.f; }
  v8f o[8] = {};

  _Float16* const pw = &smem.p[w][0];

  for (int kb0 = 0; kb0 < NNq; kb0 += 32) {
    // ---- stage K (32x128) and V^T (128x32) tiles into LDS ----
#if __has_builtin(__builtin_amdgcn_tensor_load_to_lds)
    if (w == 0) {
      // K tile: contiguous rows, stride E; lands as [key][e]
      tdm_load_2d_f16(kp + (size_t)kb0 * EEn, 0u,
                      /*tile*/ EEn, 32, /*tensor*/ EEn, 32, EEn);
      // V tile: 32 keys from each of 128 e-rows, stride N; lands as [e][key]
      tdm_load_2d_f16(vp + kb0, (unsigned)(32 * EEn * sizeof(_Float16)),
                      /*tile*/ 32, EEn, /*tensor*/ 32, EEn, NNq);
      __builtin_amdgcn_s_wait_tensorcnt(0);
    }
#else
    {
      const int tid = threadIdx.x;
      // K tile is one contiguous 8 KB block in (B,H,N,E)
      const v8h* ksrc = (const v8h*)(kp + (size_t)kb0 * EEn);
      v8h*       kdst = (v8h*)smem.k;
      for (int c = tid; c < (32 * EEn) / 8; c += 256) kdst[c] = ksrc[c];
      // V tile: 128 rows x 32 halves (64 B -> 4 chunks/row), stride N
      for (int c = tid; c < (EEn * 32) / 8; c += 256) {
        const int row = c >> 2;
        const int col = (c & 3) * 8;
        *(v8h*)(smem.v + row * 32 + col) =
            *(const v8h*)(vp + (size_t)row * NNq + kb0 + col);
      }
    }
#endif
    __syncthreads();

    // ---- S = q @ k^T  (two 16x16 column subtiles) from LDS ----
    v8f s0 = {}, s1 = {};
#pragma unroll
    for (int t = 0; t < 4; ++t) {
      v16h bk0 = frag_b_load(smem.k + 0 * EEn + t * 32, EEn);
      v16h bk1 = frag_b_load(smem.k + 16 * EEn + t * 32, EEn);
      s0 = wmma_f16(aq[t], bk0, s0);
      s1 = wmma_f16(aq[t], bk1, s1);
    }

    // ---- + dist, attention mask, online softmax; P -> LDS immediately ----
    const int   key0 = kb0 + nn, key1 = kb0 + 16 + nn;
    const float mk0  = mask[b * NNq + key0];
    const float mk1  = mask[b * NNq + key1];
    const float* distrow = dist + ((size_t)(b * NNq + qrow0 + 8 * hi)) * NNq + kb0 + nn;
#pragma unroll
    for (int r = 0; r < 8; ++r) {
      const float d0 = distrow[(size_t)r * NNq];
      const float d1 = distrow[(size_t)r * NNq + 16];
      float a0 = (mq[r] * mk0 == 0.f) ? NEGV : (s0[r] + d0);
      float a1 = (mq[r] * mk1 == 0.f) ? NEGV : (s1[r] + d1);
      float bm   = rowmax16(fmaxf(a0, a1));
      float mnew = fmaxf(mrow[r], bm);
      float alpha = __expf(mrow[r] - mnew);
      mrow[r] = mnew;
      float e0 = __expf(a0 - mnew);
      float e1 = __expf(a1 - mnew);
      lrow[r] = lrow[r] * alpha + rowsum16(e0 + e1);
#pragma unroll
      for (int c = 0; c < 8; ++c) o[c][r] *= alpha;
      // store this row's P entries now (C-frag -> row-major 16x32 in LDS)
      const int m = r + 8 * hi;
      pw[m * 32 + nn]      = (_Float16)e0;
      pw[m * 32 + 16 + nn] = (_Float16)e1;
    }
    __syncthreads();
    v16h pa = frag_a_load(pw, 32);

    // ---- O += P @ V  (K = 32 keys, 8 output column chunks of 16) ----
#pragma unroll
    for (int c = 0; c < 8; ++c) {
      v16h bv = frag_b_load(smem.v + (size_t)(c * 16) * 32, 32);
      o[c] = wmma_f16(pa, bv, o[c]);
    }
    __syncthreads();   // all waves done with K/V tiles before next overwrite
  }

  // ---- normalize and store y (B, N, H*E) ----
#pragma unroll
  for (int r = 0; r < 8; ++r) {
    const int   row = qrow0 + r + 8 * hi;
    const float inv = 1.0f / lrow[r];
    _Float16* yp = y + ((size_t)(b * NNq + row)) * HE + h * EEn;
#pragma unroll
    for (int c = 0; c < 8; ++c)
      yp[c * 16 + nn] = (_Float16)(o[c][r] * inv);
  }
}

// ---------------------------------------------------------------------------
// Kernel 3: output projection  out = (y @ Wo^T + bo) * mask
//   y: (B*N, HE) f16; woh: (E, HE) f16 row-major (== B-transposed for y@Wo^T)
// ---------------------------------------------------------------------------
__global__ void __launch_bounds__(256, 1)
out_proj_kernel(const _Float16* __restrict__ y,
                const _Float16* __restrict__ woh,
                const float* __restrict__ bo,
                const float* __restrict__ mask,
                float* __restrict__ out) {
  const int w    = threadIdx.x >> 5;
  const int lane = threadIdx.x & 31;
  const int nn   = lane & 15;
  const int hi   = lane >> 4;
  const int rowBase = (blockIdx.x * 8 + w) * 16;
  const int b  = rowBase / NNq;
  const int n0 = rowBase % NNq;

  v8f o[8] = {};
  for (int t = 0; t < HE / 32; ++t) {
    v16h a = frag_a_load(y + (size_t)rowBase * HE + t * 32, HE);
#pragma unroll
    for (int c = 0; c < 8; ++c)
      o[c] = wmma_f16(a, frag_b_load(woh + (size_t)(c * 16) * HE + t * 32, HE), o[c]);
  }

#pragma unroll
  for (int c = 0; c < 8; ++c) {
    const int   col = c * 16 + nn;
    const float bs  = bo[col];
#pragma unroll
    for (int r = 0; r < 8; ++r) {
      const int m = r + 8 * hi;
      out[((size_t)(rowBase + m)) * EEn + col] = (o[c][r] + bs) * mask[b * NNq + n0 + m];
    }
  }
}

// ---------------------------------------------------------------------------
// Host launcher
// ---------------------------------------------------------------------------
extern "C" void kernel_launch(void* const* d_in, const int* in_sizes, int n_in,
                              void* d_out, int out_size, void* d_ws, size_t ws_size,
                              hipStream_t stream) {
  (void)in_sizes; (void)n_in; (void)out_size; (void)ws_size;
  const float* x    = (const float*)d_in[0];
  const float* dist = (const float*)d_in[1];
  const float* mask = (const float*)d_in[2];
  const float* Wq   = (const float*)d_in[3];
  const float* bq   = (const float*)d_in[4];
  const float* Wk   = (const float*)d_in[5];
  const float* bk   = (const float*)d_in[6];
  const float* Wv   = (const float*)d_in[7];
  const float* bv   = (const float*)d_in[8];
  const float* Wo   = (const float*)d_in[9];
  const float* bo   = (const float*)d_in[10];
  float* out = (float*)d_out;

  // f16 workspace layout
  _Float16* ws = (_Float16*)d_ws;
  size_t off = 0;
  _Float16* xh  = ws + off; off += (size_t)BB * NNq * EEn;   // x  f16
  _Float16* wqh = ws + off; off += (size_t)HE * EEn;
  _Float16* wkh = ws + off; off += (size_t)HE * EEn;
  _Float16* wvh = ws + off; off += (size_t)HE * EEn;
  _Float16* woh = ws + off; off += (size_t)EEn * HE;
  _Float16* qws = ws + off; off += (size_t)BB * HHn * NNq * EEn;  // (B,H,N,E)
  _Float16* kws = ws + off; off += (size_t)BB * HHn * NNq * EEn;  // (B,H,N,E)
  _Float16* vtw = ws + off; off += (size_t)BB * HHn * EEn * NNq;  // (B,H,E,N)
  _Float16* yws = ws + off; off += (size_t)BB * NNq * HE;         // (B,N,HE)

  const int nx = BB * NNq * EEn;       // 1,048,576
  const int nw = HE * EEn;             // 131,072
  cvt_f32_f16<<<(nx + 255) / 256, 256, 0, stream>>>(x,  xh,  nx);
  cvt_f32_f16<<<(nw + 255) / 256, 256, 0, stream>>>(Wq, wqh, nw);
  cvt_f32_f16<<<(nw + 255) / 256, 256, 0, stream>>>(Wk, wkh, nw);
  cvt_f32_f16<<<(nw + 255) / 256, 256, 0, stream>>>(Wv, wvh, nw);
  cvt_f32_f16<<<(nw + 255) / 256, 256, 0, stream>>>(Wo, woh, nw);

  const float scale = 0.08838834764831845f;  // 128^-0.5, folded into q
  const int rowBlocks = (BB * NNq) / (16 * 8);  // 64
  qkv_proj_kernel<0><<<rowBlocks, 256, 0, stream>>>(xh, wqh, bq, mask, qws, scale);
  qkv_proj_kernel<0><<<rowBlocks, 256, 0, stream>>>(xh, wkh, bk, mask, kws, 1.0f);
  qkv_proj_kernel<1><<<rowBlocks, 256, 0, stream>>>(xh, wvh, bv, mask, vtw, 1.0f);

  attn_kernel<<<BB * HHn * (NNq / 128), 256, 0, stream>>>(qws, kws, vtw, dist, mask, yws);

  out_proj_kernel<<<rowBlocks, 256, 0, stream>>>(yws, woh, bo, mask, out);
}
